// NCEAverage_each_pre_71932112273731
// MI455X (gfx1250) — compile-verified, matrix operand order checked by reading, not run on hardware
//
#include <hip/hip_runtime.h>
#include <math.h>

typedef __attribute__((ext_vector_type(16))) _Float16 v16h;
typedef __attribute__((ext_vector_type(8)))  float    v8f;

#define BB 512
#define DD 128
#define KK 4096
#define GG 131072
#define NCOL (KK + 1)
#define N1  (BB * NCOL)
#define INV_T (1.0f / 0.07f)

// workspace layout (bytes)
#define WS_FEATN_F32 0                       // 512*128*4  = 262144
#define WS_FEATN_H   262144                  // 512*128*2  = 131072
#define WS_WNG_H     393216                  // 4096*128*2 = 1048576
#define WS_WNM_H     1441792                 // 1048576
#define WS_WPG       2490368                 // 512
#define WS_WPM       2490880                 // 512
#define WS_PARTIALS  2491392                 // 1024*4
#define WS_BCNT      2495488                 // 1024*4
#define WS_BOFF      2499584                 // 1024*4
#define WS_SCALARS   2503680                 // 256  ([0]=Ng int, [1]=invZ float)
#define WS_ROWMAP    2503936                 // 262144*4

// ---------------- feat L2 normalize: ws f32, ws f16, and tail of feature_clu_all
__global__ void featn_kernel(const float* __restrict__ feat, float* __restrict__ fo,
                             _Float16* __restrict__ fh, float* __restrict__ outTail) {
  int b = blockIdx.x, d = threadIdx.x;
  __shared__ float red[128];
  float v = feat[b * DD + d];
  red[d] = v * v; __syncthreads();
  for (int s = 64; s > 0; s >>= 1) { if (d < s) red[d] += red[d + s]; __syncthreads(); }
  float nv = v / sqrtf(red[0]);
  fo[b * DD + d] = nv;
  fh[b * DD + d] = (_Float16)nv;
  outTail[b * DD + d] = nv;
}

// ---------------- gather + normalize prototype rows (negatives -> f16, positives -> f32)
__global__ void gather_kernel(const float* __restrict__ gbm, const float* __restrict__ mate,
                              const int* __restrict__ idxng, const int* __restrict__ idxnm,
                              const int* __restrict__ idxpg, const int* __restrict__ idxpm,
                              _Float16* __restrict__ wng, _Float16* __restrict__ wnm,
                              float* __restrict__ wpg, float* __restrict__ wpm) {
  int blk = blockIdx.x, d = threadIdx.x;
  const float* src; _Float16* dsth = nullptr; float* dstf = nullptr;
  if (blk < KK)            { src = gbm  + (long)idxng[blk]      * DD; dsth = wng + (long)blk * DD; }
  else if (blk < 2 * KK)   { int k = blk - KK;
                             src = mate + (long)idxnm[k]        * DD; dsth = wnm + (long)k   * DD; }
  else if (blk == 2 * KK)  { src = gbm  + (long)idxpg[0]        * DD; dstf = wpg; }
  else                     { src = mate + (long)idxpm[0]        * DD; dstf = wpm; }
  __shared__ float red[128];
  float v = src[d];
  red[d] = v * v; __syncthreads();
  for (int s = 64; s > 0; s >>= 1) { if (d < s) red[d] += red[d + s]; __syncthreads(); }
  float nv = v / sqrtf(red[0]);
  if (dsth) dsth[d] = (_Float16)nv; else dstf[d] = nv;
}

// ---------------- positive column: exp(dot/T) into out[b*(K+1)]
__global__ void pos_kernel(const float* __restrict__ featn, const float* __restrict__ wpg,
                           const float* __restrict__ wpm, const int* __restrict__ label,
                           float* __restrict__ out) {
  int b = blockIdx.x, d = threadIdx.x;
  __shared__ float rg[128], rm[128];
  float a = featn[b * DD + d];
  rg[d] = a * wpg[d]; rm[d] = a * wpm[d]; __syncthreads();
  for (int s = 64; s > 0; s >>= 1) {
    if (d < s) { rg[d] += rg[d + s]; rm[d] += rm[d + s]; }
    __syncthreads();
  }
  if (d == 0) {
    float sel = (label[b] == 0) ? rg[0] : rm[0];
    out[(long)b * NCOL] = expf(sel * INV_T);
  }
}

// ---------------- WMMA GEMM: both banks per 16x16 tile, label-select, exp, write negatives
__global__ void gemm_kernel(const _Float16* __restrict__ fh, const _Float16* __restrict__ wng,
                            const _Float16* __restrict__ wnm, const int* __restrict__ label,
                            float* __restrict__ out) {
  int wave = (blockIdx.x << 3) | (threadIdx.x >> 5);  // 8 waves/block, 8192 tiles
  int lane = threadIdx.x & 31;
  int tm = wave >> 8;          // 0..31  (rows of feat, 16 each)
  int tn = wave & 255;         // 0..255 (cols / prototype rows, 16 each)
  int halfsel = (lane & 16) ? 8 : 0;
  const _Float16* Ar = fh  + (long)((tm << 4) + (lane & 15)) * DD;
  const _Float16* Bg = wng + (long)((tn << 4) + (lane & 15)) * DD;
  const _Float16* Bm = wnm + (long)((tn << 4) + (lane & 15)) * DD;
  v8f cg = {0.f, 0.f, 0.f, 0.f, 0.f, 0.f, 0.f, 0.f};
  v8f cm = {0.f, 0.f, 0.f, 0.f, 0.f, 0.f, 0.f, 0.f};
#pragma unroll
  for (int kc = 0; kc < 4; ++kc) {
    int k0 = kc * 32 + halfsel;
    v16h a, bg, bm;
#pragma unroll
    for (int i = 0; i < 8; ++i) {
      a[i]      = Ar[k0 + i];       a[8 + i]  = Ar[k0 + 16 + i];
      bg[i]     = Bg[k0 + i];       bg[8 + i] = Bg[k0 + 16 + i];
      bm[i]     = Bm[k0 + i];       bm[8 + i] = Bm[k0 + 16 + i];
    }
    cg = __builtin_amdgcn_wmma_f32_16x16x32_f16(false, a, false, bg, (short)0, cg, false, false);
    cm = __builtin_amdgcn_wmma_f32_16x16x32_f16(false, a, false, bm, (short)0, cm, false, false);
  }
  int col = (tn << 4) + (lane & 15);
#pragma unroll
  for (int r = 0; r < 8; ++r) {
    int m = (tm << 4) + r + halfsel;
    float sel = (label[m] == 0) ? cm[r] : cg[r];   // label==0 -> negatives from mate bank
    out[(long)m * NCOL + 1 + col] = expf(sel * INV_T);
  }
}

// ---------------- deterministic two-stage mean for Z
__global__ void reduce1_kernel(const float* __restrict__ e, float* __restrict__ partials) {
  __shared__ float red[256];
  int t = threadIdx.x;
  float s = 0.f;
  for (long i = (long)blockIdx.x * 256 + t; i < N1; i += (long)1024 * 256) s += e[i];
  red[t] = s; __syncthreads();
  for (int k = 128; k > 0; k >>= 1) { if (t < k) red[t] += red[t + k]; __syncthreads(); }
  if (t == 0) partials[blockIdx.x] = red[0];
}

__global__ void reduce2_kernel(const float* __restrict__ partials, float* __restrict__ scalars_f) {
  __shared__ float red[256];
  int t = threadIdx.x;
  float s = partials[t] + partials[t + 256] + partials[t + 512] + partials[t + 768];
  red[t] = s; __syncthreads();
  for (int k = 128; k > 0; k >>= 1) { if (t < k) red[t] += red[t + k]; __syncthreads(); }
  if (t == 0) scalars_f[1] = (float)N1 / (red[0] * (float)GG);  // 1/Z
}

__global__ void scale_kernel(float* __restrict__ e, const float* __restrict__ scalars_f) {
  float inv = scalars_f[1];
  for (long i = (long)blockIdx.x * blockDim.x + threadIdx.x; i < N1;
       i += (long)gridDim.x * blockDim.x)
    e[i] *= inv;
}

// ---------------- mask compaction: per-block counts -> scan -> ordered row map
__global__ void count_kernel(const float* __restrict__ pg, const float* __restrict__ pm,
                             int* __restrict__ bcnt) {
  int bank = blockIdx.y;
  int i = blockIdx.x * 256 + threadIdx.x;
  const float* p = bank ? pm : pg;
  float th = bank ? 0.6f : 0.5f;
  int pred = (p[i] > th) ? 1 : 0;
  __shared__ int red[256];
  int t = threadIdx.x;
  red[t] = pred; __syncthreads();
  for (int k = 128; k > 0; k >>= 1) { if (t < k) red[t] += red[t + k]; __syncthreads(); }
  if (t == 0) bcnt[bank * 512 + blockIdx.x] = red[0];
}

__global__ void scan_kernel(const int* __restrict__ bcnt, int* __restrict__ boff,
                            int* __restrict__ scalars_i) {
  __shared__ int s[1024];
  int t = threadIdx.x;
  int v = bcnt[t];
  s[t] = v; __syncthreads();
  for (int d = 1; d < 1024; d <<= 1) {
    int add = (t >= d) ? s[t - d] : 0;
    __syncthreads();
    s[t] += add;
    __syncthreads();
  }
  boff[t] = s[t] - v;                       // exclusive offsets (global: gbm blocks first)
  if (t == 511) scalars_i[0] = s[511];      // Ng
}

__global__ void compact_kernel(const float* __restrict__ pg, const float* __restrict__ pm,
                               const int* __restrict__ boff, int* __restrict__ rowmap) {
  int bank = blockIdx.y;
  int i = blockIdx.x * 256 + threadIdx.x;
  const float* p = bank ? pm : pg;
  float th = bank ? 0.6f : 0.5f;
  int pred = (p[i] > th) ? 1 : 0;
  __shared__ int s[256];
  int t = threadIdx.x;
  s[t] = pred; __syncthreads();
  for (int d = 1; d < 256; d <<= 1) {
    int add = (t >= d) ? s[t - d] : 0;
    __syncthreads();
    s[t] += add;
    __syncthreads();
  }
  if (pred) {
    int outrow = boff[bank * 512 + blockIdx.x] + s[t] - 1;
    rowmap[outrow] = i | (bank << 30);
  }
}

__global__ void rowcopy_kernel(const int* __restrict__ rowmap, const float* __restrict__ gbm,
                               const float* __restrict__ mate, float* __restrict__ out) {
  int r = blockIdx.x, d = threadIdx.x;
  int v = rowmap[r];
  const float* src = ((v >> 30) & 1) ? mate : gbm;
  long si = (long)(v & 0x3FFFFFFF) * DD;
  __shared__ float red[128];
  float x = src[si + d];
  red[d] = x * x; __syncthreads();
  for (int s = 64; s > 0; s >>= 1) { if (d < s) red[d] += red[d + s]; __syncthreads(); }
  out[(long)r * DD + d] = x / sqrtf(red[0]);
}

__global__ void gt_kernel(const int* __restrict__ label, const int* __restrict__ scalars_i,
                          float* __restrict__ out, int Ntotal) {
  int i = blockIdx.x * 256 + threadIdx.x;
  if (i >= Ntotal) return;
  int Ng = scalars_i[0];
  int Nnm = Ntotal - BB;          // Ng + Nm
  float v;
  if (i < Ng) v = 0.f;
  else if (i < Nnm) v = 1.f;
  else v = (float)label[i - Nnm];
  out[i] = v;
}

extern "C" void kernel_launch(void* const* d_in, const int* in_sizes, int n_in,
                              void* d_out, int out_size, void* d_ws, size_t ws_size,
                              hipStream_t stream) {
  const float* feat  = (const float*)d_in[0];
  const int*   label = (const int*)d_in[1];
  const float* gbm   = (const float*)d_in[5];
  const float* pg    = (const float*)d_in[6];
  const float* mate  = (const float*)d_in[7];
  const float* pm    = (const float*)d_in[8];
  const int* idxng   = (const int*)d_in[9];
  const int* idxpg   = (const int*)d_in[10];
  const int* idxnm   = (const int*)d_in[11];
  const int* idxpm   = (const int*)d_in[12];

  float* out = (float*)d_out;
  char*  ws  = (char*)d_ws;
  float*    featn = (float*)(ws + WS_FEATN_F32);
  _Float16* fh    = (_Float16*)(ws + WS_FEATN_H);
  _Float16* wng   = (_Float16*)(ws + WS_WNG_H);
  _Float16* wnm   = (_Float16*)(ws + WS_WNM_H);
  float* wpg      = (float*)(ws + WS_WPG);
  float* wpm      = (float*)(ws + WS_WPM);
  float* partials = (float*)(ws + WS_PARTIALS);
  int*   bcnt     = (int*)(ws + WS_BCNT);
  int*   boff     = (int*)(ws + WS_BOFF);
  int*   scalars_i= (int*)(ws + WS_SCALARS);
  float* scalars_f= (float*)(ws + WS_SCALARS);
  int*   rowmap   = (int*)(ws + WS_ROWMAP);

  // sizes recoverable from out_size: out_size = N1 + Ntotal*(D+1)
  int Ntotal = (out_size - N1) / (DD + 1);
  int NgNm   = Ntotal - BB;
  float* outFeat = out + N1;
  float* outGt   = out + (long)N1 + (long)Ntotal * DD;
  float* outTail = outFeat + (long)NgNm * DD;

  featn_kernel<<<BB, 128, 0, stream>>>(feat, featn, fh, outTail);
  gather_kernel<<<2 * KK + 2, 128, 0, stream>>>(gbm, mate, idxng, idxnm, idxpg, idxpm,
                                                wng, wnm, wpg, wpm);
  pos_kernel<<<BB, 128, 0, stream>>>(featn, wpg, wpm, label, out);
  gemm_kernel<<<1024, 256, 0, stream>>>(fh, wng, wnm, label, out);
  reduce1_kernel<<<1024, 256, 0, stream>>>(out, partials);
  reduce2_kernel<<<1, 256, 0, stream>>>(partials, scalars_f);
  scale_kernel<<<2048, 256, 0, stream>>>(out, scalars_f);
  count_kernel<<<dim3(512, 2), 256, 0, stream>>>(pg, pm, bcnt);
  scan_kernel<<<1, 1024, 0, stream>>>(bcnt, boff, scalars_i);
  compact_kernel<<<dim3(512, 2), 256, 0, stream>>>(pg, pm, boff, rowmap);
  if (NgNm > 0)
    rowcopy_kernel<<<NgNm, 128, 0, stream>>>(rowmap, gbm, mate, outFeat);
  gt_kernel<<<(Ntotal + 255) / 256, 256, 0, stream>>>(label, scalars_i, outGt, Ntotal);
}
